// PiscesModel_42915313222087
// MI455X (gfx1250) — matrix operand order repaired and checked by reading, not compiled, not run on hardware
//
#include <hip/hip_runtime.h>
#include <hip/hip_bf16.h>

// ---------------------------------------------------------------------------
// Model dims (fixed by the reference)
// ---------------------------------------------------------------------------
#define TT   2048
#define DD   1024
#define NH   16
#define NKVH 4
#define HDIM 64
#define FF   2048
#define NE   8
#define NL   2
#define CH   512
#define VV   32000

typedef __attribute__((ext_vector_type(16))) __bf16    v16bf;
typedef __attribute__((ext_vector_type(8)))  float     v8f;
typedef __attribute__((ext_vector_type(8)))  unsigned  v8u;
typedef __attribute__((ext_vector_type(4)))  int       v4i;

#define AS1 __attribute__((address_space(1)))
#define AS3 __attribute__((address_space(3)))

#if __has_builtin(__builtin_amdgcn_global_load_async_to_lds_b128) && \
    __has_builtin(__builtin_amdgcn_s_wait_asynccnt)
#define USE_ASYNC 1
#else
#define USE_ASYNC 0
#endif

// Pack two f32 into one bf16x2 register with a single v_perm_b32 (truncate).
__device__ __forceinline__ unsigned pkbf(float lo, float hi) {
    return __builtin_amdgcn_perm(__builtin_bit_cast(unsigned, hi),
                                 __builtin_bit_cast(unsigned, lo), 0x07060302u);
}

// 16-byte global -> LDS copy (async DMA on gfx1250 when available)
__device__ __forceinline__ void cp16(const float* g, float* l) {
#if USE_ASYNC
    __builtin_amdgcn_global_load_async_to_lds_b128(
        (AS1 v4i*)(size_t)g, (AS3 v4i*)(unsigned)(size_t)l, 0, 0);
#else
    *(float4*)l = *(const float4*)g;
#endif
}
__device__ __forceinline__ void cp16_wait_all() {
#if USE_ASYNC
    __builtin_amdgcn_s_wait_asynccnt(0);
#endif
}
// Allow the 8 just-issued (next-tile) ops to remain outstanding; ASYNC ops
// complete in order, so cnt<=8 means the previous tile's 8 ops are done.
__device__ __forceinline__ void cp16_wait_prev() {
#if USE_ASYNC
    __builtin_amdgcn_s_wait_asynccnt(8);
#endif
}

// A-fragment K index for 16-bit A 16x32 (ISA 7.12.2): vgpr j, half, lane-half kh
__device__ __forceinline__ int a_kidx(int j, int half, int kh) {
    return ((j >> 2) << 4) + 8 * kh + ((j & 3) << 1) + half;
}
// B-fragment K index for 16-bit B 32x16
__device__ __forceinline__ int b_kidx(int j, int half, int kh) {
    return ((j >> 2) << 4) + 8 * kh + ((j & 3) << 1) + half;
}

#define AROW 36      // A tile row stride: 144B, 16B-aligned, conflict-free
#define BROW 128     // B tile row stride: 512B (row-contiguous lane reads)

// ---------------------------------------------------------------------------
// Generic tiled GEMM:  C[crows[r], n] (+)= cscale[r] * sum_k A[arows[r], k]*B[k, n]
// Double-buffered f32 LDS tiles filled by global_load_async_to_lds_b128
// (exactly 8 ops/wave/k-step; OOB chunks redirected to safe dummy addresses,
// their outputs masked in the epilogue), bf16 fragments packed with v_perm_b32,
// accumulated with v_wmma_f32_16x16x32_bf16.
// Block 256 thr (8 waves), tile 128x128, wave tile 32x64. Meff read on device.
// ---------------------------------------------------------------------------
__global__ __launch_bounds__(256)
void gemm_k(const float* __restrict__ A, int lda,
            const float* __restrict__ B, int ldb,
            float* __restrict__ C, int ldc,
            int M, int N, int K,
            const int* __restrict__ arows,
            const int* __restrict__ crows,
            const float* __restrict__ cscale,
            const int* __restrict__ mcnt,
            int accumulate)
{
    int Meff = mcnt ? *mcnt : M;
    int br = blockIdx.y, bc = blockIdx.x;
    if (br * 128 >= Meff) return;   // uniform: safe before barriers

    __shared__ float sA[2][128][AROW];
    __shared__ float sB[2][32][BROW];
    __shared__ int   srow[128];

    int tid  = threadIdx.x;
    int lane = tid & 31;
    int w    = tid >> 5;
    int wr   = w >> 1;          // 0..3 -> row offset wr*32
    int wc   = w & 1;           // 0..1 -> col offset wc*64
    int l16  = lane & 15;
    int kh   = lane >> 4;

    // Stage gathered row indices once. OOB rows alias row 0 (stores masked).
    if (tid < 128) {
        int gr = br * 128 + tid;
        srow[tid] = (gr < Meff) ? (arows ? arows[gr] : gr) : 0;
    }
    __syncthreads();

    // Hoisted chunk pointers. A tile 128x32 f32 = 1024 16B chunks (8/row),
    // B tile 32x128 f32 = 1024 chunks (32/row); 4 chunks per thread each.
    const float* aptr[4]; float* adst[4];
    const float* bptr[4]; float* bdst[4];
    #pragma unroll
    for (int p = 0; p < 4; p++) {
        int ch = tid + p * 256;
        int ra = ch >> 3, ca = (ch & 7) << 2;
        aptr[p] = A + (size_t)srow[ra] * lda + ca;
        adst[p] = &sA[0][ra][ca];
        int rb = ch >> 5, cb = (ch & 31) << 2;
        int gc = bc * 128 + cb;
        // OOB column chunk -> dummy in-bounds address (B row k0); masked later.
        bptr[p] = B + ((gc < N) ? ((size_t)rb * ldb + gc) : (size_t)0);
        bdst[p] = &sB[0][rb][cb];
    }

    const v8f vzero = {};
    v8f acc[2][4];
    #pragma unroll
    for (int mi = 0; mi < 2; mi++)
        #pragma unroll
        for (int ni = 0; ni < 4; ni++) acc[mi][ni] = vzero;

    const int ABUF = 128 * AROW;
    const int BBUF = 32 * BROW;
    int nk = K >> 5;

    // Prologue: tile 0 -> buffer 0.
    #pragma unroll
    for (int p = 0; p < 4; p++) cp16(aptr[p], adst[p]);
    #pragma unroll
    for (int p = 0; p < 4; p++) cp16(bptr[p], bdst[p]);

    for (int i = 0; i < nk; i++) {
        if (i + 1 < nk) {       // issue next tile into the other buffer
            int k0n  = (i + 1) << 5;
            int bufn = (i + 1) & 1;
            #pragma unroll
            for (int p = 0; p < 4; p++) cp16(aptr[p] + k0n, adst[p] + bufn * ABUF);
            #pragma unroll
            for (int p = 0; p < 4; p++) cp16(bptr[p] + (size_t)k0n * ldb, bdst[p] + bufn * BBUF);
            cp16_wait_prev();
        } else {
            cp16_wait_all();
        }
        __syncthreads();        // tile i visible to all waves
        int buf = i & 1;
        const float (*tA)[AROW] = sA[buf];
        const float (*tB)[BROW] = sB[buf];

        v16bf bfv[4];
        #pragma unroll
        for (int ni = 0; ni < 4; ni++) {
            v8u bu;
            int n = wc * 64 + ni * 16 + l16;
            #pragma unroll
            for (int j = 0; j < 8; j++)
                bu[j] = pkbf(tB[b_kidx(j, 0, kh)][n], tB[b_kidx(j, 1, kh)][n]);
            bfv[ni] = __builtin_bit_cast(v16bf, bu);
        }
        #pragma unroll
        for (int mi = 0; mi < 2; mi++) {
            v8u au;
            int m = wr * 32 + mi * 16 + l16;
            #pragma unroll
            for (int j = 0; j < 8; j++)
                au[j] = pkbf(tA[m][a_kidx(j, 0, kh)], tA[m][a_kidx(j, 1, kh)]);
            v16bf af = __builtin_bit_cast(v16bf, au);
            #pragma unroll
            for (int ni = 0; ni < 4; ni++)
                acc[mi][ni] = __builtin_amdgcn_wmma_f32_16x16x32_bf16(
                    false, af, false, bfv[ni], (short)0, acc[mi][ni], false, false);
        }
        __syncthreads();        // all waves done reading buf before it is refilled
    }

    // Epilogue: C/D layout — vgpr j => row j + 8*(lane>>4), col lane&15
    #pragma unroll
    for (int mi = 0; mi < 2; mi++)
        #pragma unroll
        for (int ni = 0; ni < 4; ni++)
            #pragma unroll
            for (int j = 0; j < 8; j++) {
                int gr = br * 128 + wr * 32 + mi * 16 + j + 8 * kh;
                int gc = bc * 128 + wc * 64 + ni * 16 + l16;
                if (gr < Meff && gc < N) {
                    float v = acc[mi][ni][j];
                    if (cscale) v *= cscale[gr];
                    int cr = crows ? crows[gr] : gr;
                    size_t off = (size_t)cr * ldc + gc;
                    if (accumulate) C[off] += v; else C[off] = v;
                }
            }
}

// ---------------------------------------------------------------------------
// Attention: S = (Q K^T)/8 + causal mask, per (chunk c, head).
// One wave per 16x16 tile of S; grid (8, 32, NH), block 128 (4 waves).
// ---------------------------------------------------------------------------
__global__ __launch_bounds__(128)
void attn_score_k(const float* __restrict__ q, const float* __restrict__ kk,
                  float* __restrict__ S, int c)
{
    int head = blockIdx.z;
    int rowt = blockIdx.y;                       // 16-row tile 0..31
    int colt = blockIdx.x * 4 + (threadIdx.x >> 5);
    int lane = threadIdx.x & 31;
    int l16 = lane & 15, kh = lane >> 4;

    v8f acc = {};
    const float* qbase = q + ((size_t)(c * CH + rowt * 16)) * (NH * HDIM) + head * HDIM;
    const float* kbase = kk + ((size_t)(c * CH + colt * 16)) * (NKVH * HDIM) + (head >> 2) * HDIM;

    #pragma unroll
    for (int k0 = 0; k0 < HDIM; k0 += 32) {
        v8u au, bu;
        #pragma unroll
        for (int j = 0; j < 8; j++) {
            int ka = k0 + a_kidx(j, 0, kh);
            au[j] = pkbf(qbase[(size_t)l16 * (NH * HDIM) + ka],
                         qbase[(size_t)l16 * (NH * HDIM) + ka + 1]);
            int kb = k0 + b_kidx(j, 0, kh);
            bu[j] = pkbf(kbase[(size_t)l16 * (NKVH * HDIM) + kb],
                         kbase[(size_t)l16 * (NKVH * HDIM) + kb + 1]);
        }
        acc = __builtin_amdgcn_wmma_f32_16x16x32_bf16(
            false, __builtin_bit_cast(v16bf, au), false, __builtin_bit_cast(v16bf, bu),
            (short)0, acc, false, false);
    }
    #pragma unroll
    for (int j = 0; j < 8; j++) {
        int r  = rowt * 16 + j + 8 * kh;
        int cc = colt * 16 + l16;
        float sc = acc[j] * 0.125f;
        if (cc > r) sc = -1e30f;
        S[((size_t)head * CH + r) * CH + cc] = sc;
    }
}

// O = P @ V per (chunk, head). grid (1, 32, NH), block 128; wave -> 16-col tile.
__global__ __launch_bounds__(128)
void attn_av_k(const float* __restrict__ S, const float* __restrict__ v,
               float* __restrict__ o, int c)
{
    int head = blockIdx.z;
    int rowt = blockIdx.y;
    int wt   = threadIdx.x >> 5;                 // col tile 0..3
    int lane = threadIdx.x & 31;
    int l16 = lane & 15, kh = lane >> 4;

    v8f acc = {};
    const float* sbase = S + ((size_t)head * CH + rowt * 16) * CH;
    const float* vbase = v + ((size_t)(c * CH)) * (NKVH * HDIM) + (head >> 2) * HDIM + wt * 16;

    for (int k0 = 0; k0 < CH; k0 += 32) {
        v8u au, bu;
        #pragma unroll
        for (int j = 0; j < 8; j++) {
            int ka = k0 + a_kidx(j, 0, kh);
            au[j] = pkbf(sbase[(size_t)l16 * CH + ka], sbase[(size_t)l16 * CH + ka + 1]);
            int kb = k0 + b_kidx(j, 0, kh);
            bu[j] = pkbf(vbase[(size_t)kb * (NKVH * HDIM) + l16],
                         vbase[(size_t)(kb + 1) * (NKVH * HDIM) + l16]);
        }
        acc = __builtin_amdgcn_wmma_f32_16x16x32_bf16(
            false, __builtin_bit_cast(v16bf, au), false, __builtin_bit_cast(v16bf, bu),
            (short)0, acc, false, false);
    }
    #pragma unroll
    for (int j = 0; j < 8; j++) {
        int r = rowt * 16 + j + 8 * kh;
        o[((size_t)(c * CH + r)) * (NH * HDIM) + head * HDIM + wt * 16 + l16] = acc[j];
    }
}

// Row softmax over 512 entries; grid = NH*CH rows, block 128.
__global__ __launch_bounds__(128)
void softmax512_k(float* __restrict__ S)
{
    __shared__ float red[4];
    int tid = threadIdx.x, lane = tid & 31, w = tid >> 5;
    float* p = S + (size_t)blockIdx.x * CH;
    float v[4], mx = -INFINITY;
    #pragma unroll
    for (int i = 0; i < 4; i++) { v[i] = p[tid + i * 128]; mx = fmaxf(mx, v[i]); }
    for (int off = 16; off; off >>= 1) mx = fmaxf(mx, __shfl_down(mx, off, 32));
    if (lane == 0) red[w] = mx;
    __syncthreads();
    mx = fmaxf(fmaxf(red[0], red[1]), fmaxf(red[2], red[3]));
    __syncthreads();
    float s = 0.f;
    #pragma unroll
    for (int i = 0; i < 4; i++) { v[i] = __expf(v[i] - mx); s += v[i]; }
    for (int off = 16; off; off >>= 1) s += __shfl_down(s, off, 32);
    if (lane == 0) red[w] = s;
    __syncthreads();
    s = red[0] + red[1] + red[2] + red[3];
    float inv = 1.f / s;
    #pragma unroll
    for (int i = 0; i < 4; i++) p[tid + i * 128] = v[i] * inv;
}

// ---------------------------------------------------------------------------
// Elementwise / small kernels
// ---------------------------------------------------------------------------
__global__ __launch_bounds__(256)
void embed_k(const int* __restrict__ ids, const float* __restrict__ W, float* __restrict__ h)
{
    int t = blockIdx.x;
    int id = ids[t];
    for (int i = threadIdx.x; i < DD; i += 256)
        h[(size_t)t * DD + i] = W[(size_t)id * DD + i];
}

__global__ __launch_bounds__(256)
void rmsnorm_k(const float* __restrict__ x, const float* __restrict__ w,
               float* __restrict__ o)
{
    __shared__ float red[8];
    int row = blockIdx.x;
    const float* xr = x + (size_t)row * DD;
    float s = 0.f;
    for (int i = threadIdx.x; i < DD; i += 256) { float v = xr[i]; s += v * v; }
    for (int off = 16; off; off >>= 1) s += __shfl_down(s, off, 32);
    int lane = threadIdx.x & 31, wv = threadIdx.x >> 5;
    if (lane == 0) red[wv] = s;
    __syncthreads();
    if (wv == 0) {
        float t = (lane < 8) ? red[lane] : 0.f;
        for (int off = 4; off; off >>= 1) t += __shfl_down(t, off, 32);
        if (lane == 0) red[0] = t;
    }
    __syncthreads();
    float inv = rsqrtf(red[0] / (float)DD + 1e-6f);
    for (int i = threadIdx.x; i < DD; i += 256)
        o[(size_t)row * DD + i] = w[i] * xr[i] * inv;
}

__global__ void rope_k(float* __restrict__ x, int nh)
{
    int idx = blockIdx.x * blockDim.x + threadIdx.x;
    int total = TT * nh * (HDIM / 2);
    if (idx >= total) return;
    int d2  = idx & 31;
    int tmp = idx >> 5;
    int hh  = tmp % nh;
    int t   = tmp / nh;
    int pos = t & (CH - 1);                       // chunk-local position
    float inv = __expf(-((float)(2 * d2) / (float)HDIM) * 13.8155105579643f); // ln(1e6)
    float f = (float)pos * inv;
    float cs = __cosf(f), sn = __sinf(f);
    size_t base = ((size_t)t * nh + hh) * HDIM + 2 * d2;
    float x1 = x[base], x2 = x[base + 1];
    x[base]     = x1 * cs - x2 * sn;
    x[base + 1] = x1 * sn + x2 * cs;
}

__global__ void router_topk_k(const float* __restrict__ rl,
                              int* __restrict__ counts, int* __restrict__ rows,
                              float* __restrict__ wts,
                              float* __restrict__ fsum, float* __restrict__ psum)
{
    int t = blockIdx.x * blockDim.x + threadIdx.x;
    if (t >= TT) return;
    float p[NE]; float mx = -INFINITY;
    #pragma unroll
    for (int e = 0; e < NE; e++) { p[e] = rl[t * NE + e]; mx = fmaxf(mx, p[e]); }
    float s = 0.f;
    #pragma unroll
    for (int e = 0; e < NE; e++) { p[e] = __expf(p[e] - mx); s += p[e]; }
    float invs = 1.f / s;
    #pragma unroll
    for (int e = 0; e < NE; e++) p[e] *= invs;
    int i1 = 0;
    #pragma unroll
    for (int e = 1; e < NE; e++) if (p[e] > p[i1]) i1 = e;
    int i2 = (i1 == 0) ? 1 : 0;
    #pragma unroll
    for (int e = 0; e < NE; e++) if (e != i1 && p[e] > p[i2]) i2 = e;
    float w1 = p[i1], w2 = p[i2], sw = w1 + w2;
    w1 /= sw; w2 /= sw;
    int pos1 = atomicAdd(&counts[i1], 1);
    rows[i1 * TT + pos1] = t; wts[i1 * TT + pos1] = w1;
    int pos2 = atomicAdd(&counts[i2], 1);
    rows[i2 * TT + pos2] = t; wts[i2 * TT + pos2] = w2;
    atomicAdd(&fsum[i1], 1.f);
    #pragma unroll
    for (int e = 0; e < NE; e++) atomicAdd(&psum[e], p[e]);
}

__global__ void aux_accum_k(const float* __restrict__ fsum, const float* __restrict__ psum,
                            float* __restrict__ aux)
{
    if (threadIdx.x == 0 && blockIdx.x == 0) {
        float s = 0.f;
        for (int e = 0; e < NE; e++)
            s += (fsum[e] / (float)TT) * (psum[e] / (float)TT);
        *aux += (float)NE * s;
    }
}

__global__ __launch_bounds__(256)
void silumul_k(float* __restrict__ h1, const float* __restrict__ h3,
               const int* __restrict__ cnt)
{
    int n = (*cnt) * FF;
    int idx = blockIdx.x * 256 + threadIdx.x;
    if (idx < n) {
        float x = h1[idx];
        float sg = 1.f / (1.f + __expf(-x));
        h1[idx] = x * sg * h3[idx];
    }
}

__global__ __launch_bounds__(256)
void meanvec_k(const float* __restrict__ xn, float* __restrict__ mv)
{
    int d = blockIdx.x * 256 + threadIdx.x;
    if (d >= DD) return;
    float s = 0.f;
    for (int t = 0; t < TT; t++) s += xn[(size_t)t * DD + d];
    mv[d] = s / (float)TT;
}

__global__ void heads_k(const float* __restrict__ xn, const float* __restrict__ mv,
                        const float* __restrict__ task_w, const float* __restrict__ task_b,
                        const float* __restrict__ eval_w, const float* __restrict__ eval_b,
                        const float* __restrict__ aux,
                        float* __restrict__ out_task, float* __restrict__ out_eval,
                        float* __restrict__ out_aux)
{
    int tid = threadIdx.x;
    if (tid < 16) {
        float s = task_b[tid];
        for (int d = 0; d < DD; d++) s += xn[d] * task_w[d * 16 + tid];
        out_task[tid] = s;
    } else if (tid < 24) {
        int j = tid - 16;
        float s = eval_b[j];
        for (int d = 0; d < DD; d++) s += mv[d] * eval_w[d * 8 + j];
        out_eval[j] = s;
    } else if (tid == 24) {
        *out_aux = *aux;
    }
}

// ---------------------------------------------------------------------------
// Host orchestration
// ---------------------------------------------------------------------------
extern "C" void kernel_launch(void* const* d_in, const int* in_sizes, int n_in,
                              void* d_out, int out_size, void* d_ws, size_t ws_size,
                              hipStream_t stream)
{
    const int*   ids      = (const int*)  d_in[0];
    const float* embed_W  = (const float*)d_in[1];
    const float* n1       = (const float*)d_in[2];
    const float* n2       = (const float*)d_in[3];
    const float* wq       = (const float*)d_in[4];
    const float* wk       = (const float*)d_in[5];
    const float* wv       = (const float*)d_in[6];
    const float* wo       = (const float*)d_in[7];
    const float* router_w = (const float*)d_in[8];
    const float* w1       = (const float*)d_in[9];
    const float* w3       = (const float*)d_in[10];
    const float* w2       = (const float*)d_in[11];
    const float* normf    = (const float*)d_in[12];
    const float* lm_head  = (const float*)d_in[13];
    const float* task_w   = (const float*)d_in[14];
    const float* task_b   = (const float*)d_in[15];
    const float* eval_w   = (const float*)d_in[16];
    const float* eval_b   = (const float*)d_in[17];
    float* out = (float*)d_out;

    // workspace carve-up (all 4-byte elements)
    float* fp   = (float*)d_ws;
    float* h    = fp; fp += (size_t)TT * DD;       // 2M
    float* xn   = fp; fp += (size_t)TT * DD;       // 2M
    float* qb   = fp; fp += (size_t)TT * NH * HDIM;    // 2M
    float* kb   = fp; fp += (size_t)TT * NKVH * HDIM;  // 0.5M
    float* vb   = fp; fp += (size_t)TT * NKVH * HDIM;  // 0.5M
    float* ao   = fp; fp += (size_t)TT * DD;       // 2M
    float* sc   = fp; fp += (size_t)NH * CH * CH;  // 4M
    float* rl   = fp; fp += (size_t)TT * NE;
    float* h1   = fp; fp += (size_t)TT * FF;       // 4M
    float* h3   = fp; fp += (size_t)TT * FF;       // 4M
    float* mv   = fp; fp += DD;
    float* fsum = fp; fp += NE;
    float* psum = fp; fp += NE;
    float* auxb = fp; fp += 1;
    float* wts  = fp; fp += (size_t)NE * TT;
    int*   rows = (int*)fp; fp += (size_t)NE * TT;
    int*   cnts = (int*)fp; fp += NE;

    embed_k<<<TT, 256, 0, stream>>>(ids, embed_W, h);
    (void)hipMemsetAsync(auxb, 0, sizeof(float), stream);

    for (int l = 0; l < NL; l++) {
        // --- attention block ---
        rmsnorm_k<<<TT, 256, 0, stream>>>(h, n1 + (size_t)l * DD, xn);
        gemm_k<<<dim3(8, 16), 256, 0, stream>>>(xn, DD, wq + (size_t)l * DD * (NH * HDIM), NH * HDIM,
                                                qb, NH * HDIM, TT, NH * HDIM, DD,
                                                nullptr, nullptr, nullptr, nullptr, 0);
        gemm_k<<<dim3(2, 16), 256, 0, stream>>>(xn, DD, wk + (size_t)l * DD * (NKVH * HDIM), NKVH * HDIM,
                                                kb, NKVH * HDIM, TT, NKVH * HDIM, DD,
                                                nullptr, nullptr, nullptr, nullptr, 0);
        gemm_k<<<dim3(2, 16), 256, 0, stream>>>(xn, DD, wv + (size_t)l * DD * (NKVH * HDIM), NKVH * HDIM,
                                                vb, NKVH * HDIM, TT, NKVH * HDIM, DD,
                                                nullptr, nullptr, nullptr, nullptr, 0);
        rope_k<<<(TT * NH * 32 + 255) / 256, 256, 0, stream>>>(qb, NH);
        rope_k<<<(TT * NKVH * 32 + 255) / 256, 256, 0, stream>>>(kb, NKVH);

        for (int c = 0; c < TT / CH; c++) {
            attn_score_k<<<dim3(8, 32, NH), 128, 0, stream>>>(qb, kb, sc, c);
            softmax512_k<<<NH * CH, 128, 0, stream>>>(sc);
            attn_av_k<<<dim3(1, 32, NH), 128, 0, stream>>>(sc, vb, ao, c);
        }
        // h += ao @ wo
        gemm_k<<<dim3(8, 16), 256, 0, stream>>>(ao, DD, wo + (size_t)l * DD * DD, DD,
                                                h, DD, TT, DD, DD,
                                                nullptr, nullptr, nullptr, nullptr, 1);

        // --- MoE block ---
        rmsnorm_k<<<TT, 256, 0, stream>>>(h, n2 + (size_t)l * DD, xn);
        gemm_k<<<dim3(1, 16), 256, 0, stream>>>(xn, DD, router_w + (size_t)l * DD * NE, NE,
                                                rl, NE, TT, NE, DD,
                                                nullptr, nullptr, nullptr, nullptr, 0);
        (void)hipMemsetAsync(cnts, 0, NE * sizeof(int), stream);
        (void)hipMemsetAsync(fsum, 0, NE * sizeof(float), stream);
        (void)hipMemsetAsync(psum, 0, NE * sizeof(float), stream);
        router_topk_k<<<TT / 256, 256, 0, stream>>>(rl, cnts, rows, wts, fsum, psum);
        aux_accum_k<<<1, 32, 0, stream>>>(fsum, psum, auxb);

        for (int e = 0; e < NE; e++) {
            const int*   rwe = rows + (size_t)e * TT;
            const float* wte = wts  + (size_t)e * TT;
            const int*   cte = cnts + e;
            size_t wsel = (size_t)l * NE + e;
            gemm_k<<<dim3(16, 16), 256, 0, stream>>>(xn, DD, w1 + wsel * DD * FF, FF,
                                                     h1, FF, TT, FF, DD,
                                                     rwe, nullptr, nullptr, cte, 0);
            gemm_k<<<dim3(16, 16), 256, 0, stream>>>(xn, DD, w3 + wsel * DD * FF, FF,
                                                     h3, FF, TT, FF, DD,
                                                     rwe, nullptr, nullptr, cte, 0);
            silumul_k<<<(TT * FF + 255) / 256, 256, 0, stream>>>(h1, h3, cte);
            // h[token] += wt * (h1 @ w2)
            gemm_k<<<dim3(8, 16), 256, 0, stream>>>(h1, FF, w2 + wsel * FF * DD, DD,
                                                    h, DD, TT, DD, FF,
                                                    nullptr, rwe, wte, cte, 1);
        }
    }

    // --- final norm + heads ---
    rmsnorm_k<<<TT, 256, 0, stream>>>(h, normf, xn);
    gemm_k<<<dim3((VV + 127) / 128, 16), 256, 0, stream>>>(xn, DD, lm_head, VV,
                                                           out, VV, TT, VV, DD,
                                                           nullptr, nullptr, nullptr, nullptr, 0);
    meanvec_k<<<(DD + 255) / 256, 256, 0, stream>>>(xn, mv);
    size_t off = (size_t)TT * VV;
    heads_k<<<1, 32, 0, stream>>>(xn, mv, task_w, task_b, eval_w, eval_b, auxb,
                                  out + off, out + off + 16, out + off + 24);
}
